// MultiHeadSelfAttentionProjector_87488483820339
// MI455X (gfx1250) — compile-verified, matrix operand order checked by reading, not run on hardware
//
#include <hip/hip_runtime.h>
#include <hip/hip_fp16.h>

typedef __attribute__((ext_vector_type(16))) _Float16 v16h;
typedef __attribute__((ext_vector_type(8)))  _Float16 v8h;
typedef __attribute__((ext_vector_type(8)))  float    v8f;

#define BATCH 4
#define SEQ   2048
#define TDIM  768
#define VDIM  768
#define HID   512
#define NH    8
#define HD    64

// ---------------------------------------------------------------------------
// 16x32 f16 WMMA operand tile from global (A layout; B layout when the math-B
// matrix is stored transposed so its columns are contiguous rows).
// ISA layout: lane L holds row (row + L%16); halves 0..7 = K kbase+8*(L>=16)+i,
// halves 8..15 = K kbase+16+8*(L>=16)+i  => two contiguous b128 loads.
// ---------------------------------------------------------------------------
static __device__ __forceinline__ v16h load_tile16(const _Float16* __restrict__ p,
                                                   int ldk, int row, int kbase) {
  const int lane = threadIdx.x & 31;
  const int r    = row + (lane & 15);
  const int koff = kbase + ((lane >> 4) << 3);
  const _Float16* q0 = p + (size_t)r * ldk + koff;
  v8h a0 = *(const v8h*)(q0);
  v8h a1 = *(const v8h*)(q0 + 16);
  return __builtin_shufflevector(a0, a1, 0,1,2,3,4,5,6,7,8,9,10,11,12,13,14,15);
}

// Same pattern reading from an LDS-resident tile (compile-time row pitch).
template <int LDK>
static __device__ __forceinline__ v16h lds_tile16(const _Float16* base,
                                                  int row, int kbase) {
  const int lane = threadIdx.x & 31;
  const int r    = row + (lane & 15);
  const int koff = kbase + ((lane >> 4) << 3);
  const _Float16* q0 = base + r * LDK + koff;
  v8h a0 = *(const v8h*)(q0);
  v8h a1 = *(const v8h*)(q0 + 16);
  return __builtin_shufflevector(a0, a1, 0,1,2,3,4,5,6,7,8,9,10,11,12,13,14,15);
}

static __device__ __forceinline__ v8f wmma_f16(v16h a, v16h b, v8f c) {
  return __builtin_amdgcn_wmma_f32_16x16x32_f16(false, a, false, b, (short)0, c,
                                                false, false);
}

// ---------------------------------------------------------------------------
// Precision conversion helpers
// ---------------------------------------------------------------------------
__global__ void f32_to_f16_kernel(const float* __restrict__ in,
                                  _Float16* __restrict__ out, int n) {
  int i = blockIdx.x * blockDim.x + threadIdx.x;
  if (i < n) out[i] = (_Float16)in[i];
}

__global__ void transpose_f32_to_f16_kernel(const float* __restrict__ in,
                                            _Float16* __restrict__ out,
                                            int rows, int cols) {
  int i = blockIdx.x * blockDim.x + threadIdx.x;
  if (i < rows * cols) {
    int r = i / cols, c = i % cols;
    out[(size_t)c * rows + r] = (_Float16)in[i];
  }
}

// ---------------------------------------------------------------------------
// WMMA GEMM: out[M,N] = A[M,K] (f16 row-major) * Bt[N,K]^T + bias
// One wave computes a 32x64 tile (2 A-tiles x 4 B-tiles = 8 WMMA / k-step);
// a 256-thread block covers 256x64.
// mode 0: f16 out        mode 1: f16 out * scale (Q, scale=1/sqrt(HD))
// mode 2: f16 out transposed per head (V -> V^T [B*H*HD, SEQ])
// mode 3: f32 out (final projection)
// ---------------------------------------------------------------------------
__global__ __launch_bounds__(256) void gemm_wmma_kernel(
    const _Float16* __restrict__ A, const _Float16* __restrict__ Bt,
    const float* __restrict__ bias, void* __restrict__ out,
    int M, int K, int N, int mode, float scale) {
  const int lane  = threadIdx.x & 31;
  const int w     = threadIdx.x >> 5;
  const int mtile = blockIdx.x * 256 + w * 32;
  const int ntile = blockIdx.y * 64;

  v8f c00 = {}, c01 = {}, c02 = {}, c03 = {};
  v8f c10 = {}, c11 = {}, c12 = {}, c13 = {};
  for (int k = 0; k < K; k += 32) {
    v16h a0 = load_tile16(A,  K, mtile,      k);
    v16h a1 = load_tile16(A,  K, mtile + 16, k);
    v16h b0 = load_tile16(Bt, K, ntile +  0, k);
    v16h b1 = load_tile16(Bt, K, ntile + 16, k);
    v16h b2 = load_tile16(Bt, K, ntile + 32, k);
    v16h b3 = load_tile16(Bt, K, ntile + 48, k);
    c00 = wmma_f16(a0, b0, c00);
    c01 = wmma_f16(a0, b1, c01);
    c02 = wmma_f16(a0, b2, c02);
    c03 = wmma_f16(a0, b3, c03);
    c10 = wmma_f16(a1, b0, c10);
    c11 = wmma_f16(a1, b1, c11);
    c12 = wmma_f16(a1, b2, c12);
    c13 = wmma_f16(a1, b3, c13);
  }

  const int nn = lane & 15, hi = lane >> 4;
#pragma unroll
  for (int half = 0; half < 2; ++half) {
#pragma unroll
    for (int t = 0; t < 4; ++t) {
      v8f cc;
      if (half == 0) cc = (t == 0) ? c00 : (t == 1) ? c01 : (t == 2) ? c02 : c03;
      else           cc = (t == 0) ? c10 : (t == 1) ? c11 : (t == 2) ? c12 : c13;
      const int n    = ntile + t * 16 + nn;
      const float bv = bias[n];
#pragma unroll
      for (int r = 0; r < 8; ++r) {
        const int m   = mtile + half * 16 + r + 8 * hi;
        const float v = (cc[r] + bv) * scale;
        if (mode == 2) {
          const int b = m / SEQ, s = m % SEQ;
          const int h = n / HD,  d = n % HD;
          ((_Float16*)out)[((size_t)((b * NH + h) * HD + d)) * SEQ + s] = (_Float16)v;
        } else if (mode == 3) {
          ((float*)out)[(size_t)m * N + n] = v;
        } else {
          ((_Float16*)out)[(size_t)m * N + n] = (_Float16)v;
        }
      }
    }
  }
}

// ---------------------------------------------------------------------------
// Flash attention: block = 8 waves, all on one (b,h); wave = 16-query tile.
// Per 32-kv step: the block cooperatively stages the K tile (32x64) and the
// V^T tile (64x32) in LDS (each wave previously fetched them 8x redundantly
// from global), then each wave does 4 WMMAs for Q*K^T, online softmax via
// 16-lane shuffle reductions, P transposed D->A layout through LDS, and
// 4 WMMAs for P*V.
// ---------------------------------------------------------------------------
__global__ __launch_bounds__(256) void flash_attn_kernel(
    const _Float16* __restrict__ Q, const _Float16* __restrict__ Kc,
    const _Float16* __restrict__ Vt, _Float16* __restrict__ O) {
  __shared__ __align__(16) _Float16 ldsP[8][2][16][16];  // per-wave P staging
  __shared__ __align__(16) _Float16 ldsK[32][64];        // K tile  (kv x d)
  __shared__ __align__(16) _Float16 ldsV[64][32];        // V^T tile (d x kv)

  const int tid     = threadIdx.x;
  const int lane    = tid & 31;
  const int w       = tid >> 5;
  const int qblocks = SEQ / 128;
  const int bh      = blockIdx.x / qblocks;
  const int qblk    = blockIdx.x % qblocks;
  const int b       = bh / NH, h = bh % NH;
  const int qbase   = qblk * 128 + w * 16;
  const int nn = lane & 15, hi = lane >> 4;

  const _Float16* Qp = Q  + (size_t)(b * SEQ + qbase) * HID + h * HD;
  const _Float16* Kp = Kc + (size_t)(b * SEQ) * HID + h * HD;
  const _Float16* Vp = Vt + (size_t)(bh * HD) * SEQ;

  const v16h qa0 = load_tile16(Qp, HID, 0, 0);
  const v16h qa1 = load_tile16(Qp, HID, 0, 32);

  v8f o0 = {}, o1 = {}, o2 = {}, o3 = {};
  float mrow[8], lrow[8];
#pragma unroll
  for (int r = 0; r < 8; ++r) { mrow[r] = -1e30f; lrow[r] = 0.f; }

  // cooperative staging coordinates (fixed per thread)
  const int kR = tid >> 3, kC = (tid & 7) * 8;   // 32 rows x 8 chunks
  const int vR = tid >> 2, vC = (tid & 3) * 8;   // 64 rows x 4 chunks

  for (int kv = 0; kv < SEQ; kv += 32) {
    __syncthreads();  // previous step's LDS reads complete
    *(v8h*)&ldsK[kR][kC] = *(const v8h*)(Kp + (size_t)(kv + kR) * HID + kC);
    *(v8h*)&ldsV[vR][vC] = *(const v8h*)(Vp + (size_t)vR * SEQ + kv + vC);
    __syncthreads();  // tiles visible to all waves

    // scores S[16 x 32] = Qs * K^T (B-operand columns == K rows)
    v16h kb00 = lds_tile16<64>(&ldsK[0][0],  0,  0);
    v16h kb01 = lds_tile16<64>(&ldsK[0][0],  0, 32);
    v16h kb10 = lds_tile16<64>(&ldsK[0][0], 16,  0);
    v16h kb11 = lds_tile16<64>(&ldsK[0][0], 16, 32);
    v8f s0 = {}, s1 = {};
    s0 = wmma_f16(qa0, kb00, s0);
    s0 = wmma_f16(qa1, kb01, s0);
    s1 = wmma_f16(qa0, kb10, s1);
    s1 = wmma_f16(qa1, kb11, s1);

    // online softmax; row M = r + 8*hi lives across one 16-lane half-group
#pragma unroll
    for (int r = 0; r < 8; ++r) {
      float mx = fmaxf(s0[r], s1[r]);
      mx = fmaxf(mx, __shfl_xor(mx, 1, 32));
      mx = fmaxf(mx, __shfl_xor(mx, 2, 32));
      mx = fmaxf(mx, __shfl_xor(mx, 4, 32));
      mx = fmaxf(mx, __shfl_xor(mx, 8, 32));
      const float mnew = fmaxf(mrow[r], mx);
      const float corr = __expf(mrow[r] - mnew);
      mrow[r] = mnew;
      const float p0 = __expf(s0[r] - mnew);
      const float p1 = __expf(s1[r] - mnew);
      float rs = p0 + p1;
      rs += __shfl_xor(rs, 1, 32);
      rs += __shfl_xor(rs, 2, 32);
      rs += __shfl_xor(rs, 4, 32);
      rs += __shfl_xor(rs, 8, 32);
      lrow[r] = lrow[r] * corr + rs;
      o0[r] *= corr; o1[r] *= corr; o2[r] *= corr; o3[r] *= corr;
      const int m = r + 8 * hi;                 // D layout -> LDS [m][n]
      ldsP[w][0][m][nn] = (_Float16)p0;
      ldsP[w][1][m][nn] = (_Float16)p1;
    }
    // same-wave DS ops are in-order; fence compiler + counters explicitly
    asm volatile("s_wait_dscnt 0" ::: "memory");

    // reload P in A layout: lane L -> row L%16, chunk at k = 8*(L>=16)
    v8h pl0 = *(const v8h*)&ldsP[w][0][nn][hi * 8];
    v8h pl1 = *(const v8h*)&ldsP[w][1][nn][hi * 8];
    asm volatile("" ::: "memory");
    v16h pa = __builtin_shufflevector(pl0, pl1,
                                      0,1,2,3,4,5,6,7,8,9,10,11,12,13,14,15);

    // P * V : B-operand columns of V == rows of V^T tile
    v16h vb0 = lds_tile16<32>(&ldsV[0][0],  0, 0);
    v16h vb1 = lds_tile16<32>(&ldsV[0][0], 16, 0);
    v16h vb2 = lds_tile16<32>(&ldsV[0][0], 32, 0);
    v16h vb3 = lds_tile16<32>(&ldsV[0][0], 48, 0);
    o0 = wmma_f16(pa, vb0, o0);
    o1 = wmma_f16(pa, vb1, o1);
    o2 = wmma_f16(pa, vb2, o2);
    o3 = wmma_f16(pa, vb3, o3);
  }

  // normalize and store attended (row-major [B*S, HID], head-interleaved)
#pragma unroll
  for (int r = 0; r < 8; ++r) {
    const float inv = 1.0f / lrow[r];
    const int m = qbase + r + 8 * hi;
    const size_t base = (size_t)(b * SEQ + m) * HID + h * HD + nn;
    O[base +  0] = (_Float16)(o0[r] * inv);
    O[base + 16] = (_Float16)(o1[r] * inv);
    O[base + 32] = (_Float16)(o2[r] * inv);
    O[base + 48] = (_Float16)(o3[r] * inv);
  }
}

// ---------------------------------------------------------------------------
extern "C" void kernel_launch(void* const* d_in, const int* in_sizes, int n_in,
                              void* d_out, int out_size, void* d_ws, size_t ws_size,
                              hipStream_t stream) {
  (void)in_sizes; (void)n_in; (void)out_size; (void)ws_size;
  const float* X  = (const float*)d_in[0];
  const float* Wq = (const float*)d_in[1];
  const float* bq = (const float*)d_in[2];
  const float* Wk = (const float*)d_in[3];
  const float* bk = (const float*)d_in[4];
  const float* Wv = (const float*)d_in[5];
  const float* bv = (const float*)d_in[6];
  const float* Wo = (const float*)d_in[7];
  const float* bo = (const float*)d_in[8];

  const int M = BATCH * SEQ;  // 8192
  char* ws = (char*)d_ws;
  _Float16* Xh  = (_Float16*)ws; ws += (size_t)M * TDIM * 2;
  _Float16* Wqt = (_Float16*)ws; ws += (size_t)HID * TDIM * 2;
  _Float16* Wkt = (_Float16*)ws; ws += (size_t)HID * TDIM * 2;
  _Float16* Wvt = (_Float16*)ws; ws += (size_t)HID * TDIM * 2;
  _Float16* Wot = (_Float16*)ws; ws += (size_t)VDIM * HID * 2;
  _Float16* Qh  = (_Float16*)ws; ws += (size_t)M * HID * 2;
  _Float16* Kh  = (_Float16*)ws; ws += (size_t)M * HID * 2;
  _Float16* Vth = (_Float16*)ws; ws += (size_t)M * HID * 2;  // [B*H*HD, SEQ]
  _Float16* At  = (_Float16*)ws; ws += (size_t)M * HID * 2;

  int n = M * TDIM;
  f32_to_f16_kernel<<<(n + 255) / 256, 256, 0, stream>>>(X, Xh, n);
  n = TDIM * HID;
  transpose_f32_to_f16_kernel<<<(n + 255) / 256, 256, 0, stream>>>(Wq, Wqt, TDIM, HID);
  transpose_f32_to_f16_kernel<<<(n + 255) / 256, 256, 0, stream>>>(Wk, Wkt, TDIM, HID);
  transpose_f32_to_f16_kernel<<<(n + 255) / 256, 256, 0, stream>>>(Wv, Wvt, TDIM, HID);
  n = HID * VDIM;
  transpose_f32_to_f16_kernel<<<(n + 255) / 256, 256, 0, stream>>>(Wo, Wot, HID, VDIM);

  dim3 gqkv(M / 256, HID / 64);
  gemm_wmma_kernel<<<gqkv, 256, 0, stream>>>(Xh, Wqt, bq, Qh,  M, TDIM, HID, 1, 0.125f);
  gemm_wmma_kernel<<<gqkv, 256, 0, stream>>>(Xh, Wkt, bk, Kh,  M, TDIM, HID, 0, 1.0f);
  gemm_wmma_kernel<<<gqkv, 256, 0, stream>>>(Xh, Wvt, bv, Vth, M, TDIM, HID, 2, 1.0f);

  flash_attn_kernel<<<BATCH * NH * (SEQ / 128), 256, 0, stream>>>(Qh, Kh, Vth, At);

  dim3 gout(M / 256, VDIM / 64);
  gemm_wmma_kernel<<<gout, 256, 0, stream>>>(At, Wot, bo, d_out, M, HID, VDIM, 3, 1.0f);
}
